// GAT_62251255988402
// MI455X (gfx1250) — compile-verified, matrix operand order checked by reading, not run on hardware
//
#include <hip/hip_runtime.h>

// ---------------------------------------------------------------------------
// GAT (3-layer, 4 heads x 32) for MI455X / gfx1250, wave32 + WMMA bf16.
//
// N=50000, E=800000, Etot=E+N. Per layer:
//   1) h = act @ W          -> v_wmma_f32_16x16x32_bf16, LDS-tiled
//   2) as/ad = <h, a_src/dst> per (node, head)
//   3) edge pass A: m[dst] = max(leaky_relu(as[src]+ad[dst]))  (u32-keyed atomicMax)
//   4) edge pass B (fused): ex = exp(e-m);  denom[dst]+=ex;  agg[dst]+=h[src]*ex
//   5) finalize: y = agg/denom + bias, accumulate BN sums
//   6) BN + ELU -> next activation
// ---------------------------------------------------------------------------

typedef __bf16 bf16;
typedef bf16  v8bf  __attribute__((ext_vector_type(8)));
typedef bf16  v16bf __attribute__((ext_vector_type(16)));
typedef float v8f   __attribute__((ext_vector_type(8)));

// --------------------------- monotonic float key ---------------------------
__device__ __forceinline__ unsigned fkey(float f) {
    unsigned u = __float_as_uint(f);
    return (u & 0x80000000u) ? ~u : (u | 0x80000000u);
}
__device__ __forceinline__ float unkey(unsigned k) {
    return __uint_as_float((k & 0x80000000u) ? (k ^ 0x80000000u) : ~k);
}

// ------------------------------- zero fill ---------------------------------
__global__ void k_zero(unsigned* __restrict__ p, long long cnt) {
    long long i = (long long)blockIdx.x * blockDim.x + threadIdx.x;
    long long stride = (long long)gridDim.x * blockDim.x;
    for (; i < cnt; i += stride) p[i] = 0u;
}

// ------------------------------ WMMA GEMM ----------------------------------
// A [n, K] f32 row-major, W [K, M] f32 row-major, out [n, M] f32 (+ optional bias).
// Block: 128 threads = 4 waves; each wave owns a 16-row strip; block = 64 rows.
// LDS: W transposed [M][K] bf16, A tile [64][K] bf16.
template<int K, int M>
__global__ __launch_bounds__(128) void k_gemm(const float* __restrict__ A,
                                              const float* __restrict__ W,
                                              const float* __restrict__ bias,
                                              float* __restrict__ Hout, int n) {
    __shared__ bf16 sW[M * K];   // [m][k]
    __shared__ bf16 sA[64 * K];  // [r][k]
    const int tid = threadIdx.x;

    for (int i = tid; i < K * M; i += 128) {
        int k = i / M, m = i % M;
        sW[m * K + k] = (bf16)W[i];
    }
    const int row0 = blockIdx.x * 64;
    for (int i = tid; i < 64 * K; i += 128) {
        int r = i / K, k = i % K;
        int gr = row0 + r;
        sA[i] = (bf16)(gr < n ? A[(long long)gr * K + k] : 0.0f);
    }
    __syncthreads();

    const int wave = tid >> 5;
    const int lane = tid & 31;
    const int lrow = lane & 15;   // row (A) / col (B) within tile
    const int hi   = lane >> 4;   // K-half selector per ISA 16-bit layout

    constexpr int NT = M / 16;
    v8f acc[NT] = {};

    const bf16* aRow = &sA[(wave * 16 + lrow) * K];
#pragma unroll
    for (int ks = 0; ks < K; ks += 32) {
        // A frag: lanes 0-15 hold K = ks+{0..7,16..23}; lanes 16-31: ks+{8..15,24..31}
        const v8bf* ap = (const v8bf*)(aRow + ks + hi * 8);
        v8bf a_lo = ap[0], a_hi = ap[2];
        v16bf afrag = __builtin_shufflevector(a_lo, a_hi,
            0,1,2,3,4,5,6,7,8,9,10,11,12,13,14,15);
#pragma unroll
        for (int t = 0; t < NT; ++t) {
            const v8bf* bp = (const v8bf*)(&sW[(t * 16 + lrow) * K] + ks + hi * 8);
            v8bf b_lo = bp[0], b_hi = bp[2];
            v16bf bfrag = __builtin_shufflevector(b_lo, b_hi,
                0,1,2,3,4,5,6,7,8,9,10,11,12,13,14,15);
            acc[t] = __builtin_amdgcn_wmma_f32_16x16x32_bf16(
                false, afrag, false, bfrag, (short)0, acc[t], false, false);
        }
    }

    // D layout: VGPR r, lanes 0-15 -> M=r, lanes 16-31 -> M=8+r; N = lane%16
    const int gr_base = row0 + wave * 16 + hi * 8;
#pragma unroll
    for (int r = 0; r < 8; ++r) {
        int gr = gr_base + r;
        if (gr < n) {
#pragma unroll
            for (int t = 0; t < NT; ++t) {
                int col = t * 16 + lrow;
                float b = bias ? bias[col] : 0.0f;
                Hout[(long long)gr * M + col] = acc[t][r] + b;
            }
        }
    }
}

// --------------------- per-(node, head) attention dots ---------------------
__global__ void k_att(const float* __restrict__ h, const float* __restrict__ a_src,
                      const float* __restrict__ a_dst, float* __restrict__ as_,
                      float* __restrict__ ad_, int n, int H, int C) {
    int i = blockIdx.x * blockDim.x + threadIdx.x;   // node*H + head
    if (i >= n * H) return;
    int hd = i % H;
    const float* hp = h + (long long)i * C;
    const float* s  = a_src + hd * C;
    const float* d  = a_dst + hd * C;
    float vs = 0.f, vd = 0.f;
    for (int c = 0; c < C; ++c) { float x = hp[c]; vs += x * s[c]; vd += x * d[c]; }
    as_[i] = vs; ad_[i] = vd;
}

// --------------------- edge pass A: segment max (keyed) ---------------------
__global__ void k_edge_max(const int* __restrict__ ei, const float* __restrict__ as_,
                           const float* __restrict__ ad_, unsigned* __restrict__ mkey,
                           int E_, int n, int H) {
    long long i = (long long)blockIdx.x * blockDim.x + threadIdx.x;
    long long tot = (long long)(E_ + n) * H;
    if (i >= tot) return;
    int e  = (int)(i / H), hd = (int)(i % H);
    int s  = e < E_ ? ei[e]      : (e - E_);
    int d  = e < E_ ? ei[E_ + e] : (e - E_);
    float x  = as_[s * H + hd] + ad_[d * H + hd];
    float lr = x > 0.f ? x : 0.2f * x;
    atomicMax(&mkey[d * H + hd], fkey(lr));
}

// -------- edge pass B (fused): exp, denom += ex, agg += h[src]*ex ----------
// One wave32 per edge: coalesced 128B row gather, per-lane f32 atomics.
template<int H, int C>
__global__ __launch_bounds__(256) void k_edge_agg(
        const int* __restrict__ ei, const float* __restrict__ as_,
        const float* __restrict__ ad_, const unsigned* __restrict__ mkey,
        const float* __restrict__ h, float* __restrict__ agg,
        float* __restrict__ denom, int E_, int n) {
    constexpr int MH = H * C;
    int e    = blockIdx.x * (blockDim.x >> 5) + (threadIdx.x >> 5);
    int lane = threadIdx.x & 31;
    if (e >= E_ + n) return;
    int s = e < E_ ? ei[e]      : (e - E_);
    int d = e < E_ ? ei[E_ + e] : (e - E_);

    float ex[H];
#pragma unroll
    for (int hd = 0; hd < H; ++hd) {
        float x  = as_[s * H + hd] + ad_[d * H + hd];   // SGPR-broadcast loads
        float lr = x > 0.f ? x : 0.2f * x;
        float m  = unkey(mkey[d * H + hd]);
        ex[hd]   = __expf(lr - m);
    }
    if (lane < H) atomicAdd(&denom[d * H + lane], ex[lane]);
#pragma unroll
    for (int i = 0; i < MH / 32; ++i) {
        int c  = lane + 32 * i;
        int hd = c / C;
        atomicAdd(&agg[(long long)d * MH + c], h[(long long)s * MH + c] * ex[hd]);
    }
}

// ----- normalize by denom, add bias, accumulate BN sums (per channel) ------
template<int H, int C>
__global__ __launch_bounds__(128) void k_finalize(float* __restrict__ agg,
                                                  const float* __restrict__ denom,
                                                  const float* __restrict__ bias,
                                                  float* __restrict__ stats, int n) {
    constexpr int MH = H * C;
    int c  = threadIdx.x;           // blockDim.x == MH
    int hd = c / C;
    float s = 0.f, sq = 0.f;
    for (int node = blockIdx.x; node < n; node += gridDim.x) {
        float den = denom[node * H + hd] + 1e-16f;
        long long idx = (long long)node * MH + c;
        float y = agg[idx] / den + bias[c];
        agg[idx] = y;
        s += y; sq += y * y;
    }
    atomicAdd(&stats[c], s);
    atomicAdd(&stats[MH + c], sq);
}

// --------------------------- BatchNorm + ELU -------------------------------
__global__ void k_bn_elu(const float* __restrict__ yin, const float* __restrict__ stats,
                         const float* __restrict__ g, const float* __restrict__ be,
                         float* __restrict__ out, int n, int MH) {
    long long i = (long long)blockIdx.x * blockDim.x + threadIdx.x;
    if (i >= (long long)n * MH) return;
    int c = (int)(i % MH);
    float inv_n = 1.0f / (float)n;
    float mean = stats[c] * inv_n;
    float var  = stats[MH + c] * inv_n - mean * mean;
    float y = (yin[i] - mean) * rsqrtf(var + 1e-5f) * g[c] + be[c];
    out[i] = y > 0.f ? y : (__expf(y) - 1.0f);
}

// ---------------------------------------------------------------------------
extern "C" void kernel_launch(void* const* d_in, const int* in_sizes, int n_in,
                              void* d_out, int out_size, void* d_ws, size_t ws_size,
                              hipStream_t stream) {
    const float* x      = (const float*)d_in[0];
    const int*   ei     = (const int*)  d_in[1];
    const float* W1     = (const float*)d_in[2];
    const float* a1s    = (const float*)d_in[3];
    const float* a1d    = (const float*)d_in[4];
    const float* b1     = (const float*)d_in[5];
    const float* g1     = (const float*)d_in[6];
    const float* be1    = (const float*)d_in[7];
    const float* W2     = (const float*)d_in[8];
    const float* a2s    = (const float*)d_in[9];
    const float* a2d    = (const float*)d_in[10];
    const float* b2     = (const float*)d_in[11];
    const float* g2     = (const float*)d_in[12];
    const float* be2    = (const float*)d_in[13];
    const float* W3     = (const float*)d_in[14];
    const float* a3s    = (const float*)d_in[15];
    const float* a3d    = (const float*)d_in[16];
    const float* b3     = (const float*)d_in[17];
    const float* g3     = (const float*)d_in[18];
    const float* be3    = (const float*)d_in[19];
    const float* Wo     = (const float*)d_in[20];
    const float* bo     = (const float*)d_in[21];

    const int N_ = in_sizes[0] / 128;    // 50000
    const int E_ = in_sizes[1] / 2;      // 800000
    const int Etot = E_ + N_;

    // ---- workspace layout (floats) ----
    float*    bufH   = (float*)d_ws;                       // [N,128]
    float*    bufAgg = bufH   + (size_t)N_ * 128;          // [N,128]
    float*    bufAct = bufAgg + (size_t)N_ * 128;          // [N,128]
    float*    asb    = bufAct + (size_t)N_ * 128;          // [N,4]
    float*    adb    = asb    + (size_t)N_ * 4;            // [N,4]
    float*    denom  = adb    + (size_t)N_ * 4;            // [N,4]
    unsigned* mkey   = (unsigned*)(denom + (size_t)N_ * 4);// [N,4]
    float*    stats  = (float*)(mkey + (size_t)N_ * 4);    // [256]

    const int gemmBlocks = (N_ + 63) / 64;
    const int edgeWBlocks = (Etot + 7) / 8;                // 8 waves/block

    // =========================== Layer 1 (H=4, C=32) ==========================
    k_gemm<128, 128><<<gemmBlocks, 128, 0, stream>>>(x, W1, nullptr, bufH, N_);
    k_att<<<(N_ * 4 + 255) / 256, 256, 0, stream>>>(bufH, a1s, a1d, asb, adb, N_, 4, 32);
    k_zero<<<2048, 256, 0, stream>>>((unsigned*)denom, (long long)N_ * 8 + 256);
    k_zero<<<4096, 256, 0, stream>>>((unsigned*)bufAgg, (long long)N_ * 128);
    k_edge_max<<<((long long)Etot * 4 + 255) / 256, 256, 0, stream>>>(ei, asb, adb, mkey, E_, N_, 4);
    k_edge_agg<4, 32><<<edgeWBlocks, 256, 0, stream>>>(ei, asb, adb, mkey, bufH, bufAgg, denom, E_, N_);
    k_finalize<4, 32><<<1024, 128, 0, stream>>>(bufAgg, denom, b1, stats, N_);
    k_bn_elu<<<((long long)N_ * 128 + 255) / 256, 256, 0, stream>>>(bufAgg, stats, g1, be1, bufAct, N_, 128);

    // =========================== Layer 2 (H=4, C=32) ==========================
    k_gemm<128, 128><<<gemmBlocks, 128, 0, stream>>>(bufAct, W2, nullptr, bufH, N_);
    k_att<<<(N_ * 4 + 255) / 256, 256, 0, stream>>>(bufH, a2s, a2d, asb, adb, N_, 4, 32);
    k_zero<<<2048, 256, 0, stream>>>((unsigned*)denom, (long long)N_ * 8 + 256);
    k_zero<<<4096, 256, 0, stream>>>((unsigned*)bufAgg, (long long)N_ * 128);
    k_edge_max<<<((long long)Etot * 4 + 255) / 256, 256, 0, stream>>>(ei, asb, adb, mkey, E_, N_, 4);
    k_edge_agg<4, 32><<<edgeWBlocks, 256, 0, stream>>>(ei, asb, adb, mkey, bufH, bufAgg, denom, E_, N_);
    k_finalize<4, 32><<<1024, 128, 0, stream>>>(bufAgg, denom, b2, stats, N_);
    k_bn_elu<<<((long long)N_ * 128 + 255) / 256, 256, 0, stream>>>(bufAgg, stats, g2, be2, bufAct, N_, 128);

    // =========================== Layer 3 (H=1, C=32) ==========================
    k_gemm<128, 32><<<gemmBlocks, 128, 0, stream>>>(bufAct, W3, nullptr, bufH, N_);
    k_att<<<(N_ + 255) / 256, 256, 0, stream>>>(bufH, a3s, a3d, asb, adb, N_, 1, 32);
    k_zero<<<2048, 256, 0, stream>>>((unsigned*)denom, (long long)N_ * 8 + 256);
    k_zero<<<2048, 256, 0, stream>>>((unsigned*)bufAgg, (long long)N_ * 32);
    k_edge_max<<<((long long)Etot + 255) / 256, 256, 0, stream>>>(ei, asb, adb, mkey, E_, N_, 1);
    k_edge_agg<1, 32><<<edgeWBlocks, 256, 0, stream>>>(ei, asb, adb, mkey, bufH, bufAgg, denom, E_, N_);
    k_finalize<1, 32><<<1024, 32, 0, stream>>>(bufAgg, denom, b3, stats, N_);
    k_bn_elu<<<((long long)N_ * 32 + 255) / 256, 256, 0, stream>>>(bufAgg, stats, g3, be3, bufAct, N_, 32);

    // ======================= Output projection [N,16] =========================
    k_gemm<32, 16><<<gemmBlocks, 128, 0, stream>>>(bufAct, Wo, bo, (float*)d_out, N_);
}
